// PLSTM_Layer_48696339202289
// MI455X (gfx1250) — matrix operand order repaired and checked by reading, not compiled, not run on hardware
//
#include <hip/hip_runtime.h>
#include <hip/hip_bf16.h>

// ---------------- problem constants ----------------
#define S_LEN 2048
#define BATCH 32
#define INDIM 512
#define HIDN  512
#define G4    (4 * HIDN)   // 2048 gate columns
#define NWG   16           // persistent workgroups in the recurrent kernel

// ---------------- types ----------------
typedef __bf16 bf16_t;
typedef bf16_t v16bf __attribute__((ext_vector_type(16)));
typedef float  v8f   __attribute__((ext_vector_type(8)));
typedef unsigned int v4u __attribute__((ext_vector_type(4)));
typedef int          v8i __attribute__((ext_vector_type(8)));
typedef int          v4i __attribute__((ext_vector_type(4)));

union BFrag {
    v16bf          v;
    uint4          q[2];
    unsigned short us[16];
};

union Pack8 {
    uint4          q;
    unsigned short us[8];
};

// hardware f32 -> bf16 (RNE) via native cast
__device__ __forceinline__ unsigned short f2bf(float f) {
    __bf16 b = (__bf16)f;
    return __builtin_bit_cast(unsigned short, b);
}

__device__ __forceinline__ v8f zero8() {
    v8f z = {0.f, 0.f, 0.f, 0.f, 0.f, 0.f, 0.f, 0.f};
    return z;
}

__device__ __forceinline__ float sigf(float x) {
    return 1.0f / (1.0f + __expf(-x));
}

// generic -> LDS address-space offset (for TDM descriptors)
#define LDS_OFFSET(p) ((unsigned)(unsigned long long)(__attribute__((address_space(3))) void*)(p))

// ---------------- TDM: tensor load global -> LDS with row padding ----------------
// Moves tile (tile_d0 elems x tile_d1 rows [x tile_d2 blocks]) of 2-byte elements.
// LDS pad: 4 DWORDs (16B = 8 elems) inserted after every 256 DWORDs (1024B = 512
// elems) -> LDS row pitch 520 bf16, matching the bank-conflict-free layout.
__device__ __forceinline__ void tdm_load_bf16(
    unsigned lds_off, const void* gaddr,
    unsigned tensor_d0, unsigned tensor_d1, unsigned tensor_d2,
    unsigned tile_d0, unsigned tile_d1, unsigned tile_d2,
    unsigned long long stride0, unsigned long long stride1) {
    unsigned long long ga = (unsigned long long)gaddr;
    v4u g0;
    g0[0] = 1u;                                            // count=1 (valid user D#)
    g0[1] = lds_off;                                       // lds_addr
    g0[2] = (unsigned)ga;                                  // global_addr[31:0]
    g0[3] = (unsigned)((ga >> 32) & 0x01FFFFFFu) | (2u << 30); // addr[56:32] | type=2
    v8i g1;
    g1[0] = (int)((1u << 16) |                             // data_size = 2 bytes
                  (1u << 20) |                             // pad_enable
                  (7u << 22) |                             // pad_interval: 256 DW
                  (3u << 25));                             // pad_amount: 4 DW
    g1[1] = (int)((tensor_d0 & 0xFFFFu) << 16);            // tensor_dim0[15:0]
    g1[2] = (int)((tensor_d0 >> 16) | ((tensor_d1 & 0xFFFFu) << 16));
    g1[3] = (int)((tensor_d1 >> 16) | (tile_d0 << 16));    // tile_dim0
    g1[4] = (int)(tile_d1 | (tile_d2 << 16));              // tile_dim1 | tile_dim2
    g1[5] = (int)(unsigned)(stride0 & 0xFFFFFFFFu);        // tensor_dim0_stride lo
    g1[6] = (int)((unsigned)(stride0 >> 32) | ((unsigned)(stride1 & 0xFFFFu) << 16));
    g1[7] = (int)(unsigned)((stride1 >> 16) & 0xFFFFFFFFu);
    v4i g2;
    g2[0] = (int)tensor_d2; g2[1] = 0; g2[2] = 0; g2[3] = 0;
    v4i g3;
    g3[0] = 0; g3[1] = 0; g3[2] = 0; g3[3] = 0;
    v8i g4;
    g4[0] = 0; g4[1] = 0; g4[2] = 0; g4[3] = 0;
    g4[4] = 0; g4[5] = 0; g4[6] = 0; g4[7] = 0;
    __builtin_amdgcn_tensor_load_to_lds(g0, g1, g2, g3, g4, 0);
}

// ---------------- grid barrier (16 co-resident WGs) ----------------
__device__ __forceinline__ void gridBarrier(unsigned int* sync) {
    __syncthreads();
    if (threadIdx.x == 0) {
        unsigned g = __hip_atomic_load(&sync[1], __ATOMIC_ACQUIRE, __HIP_MEMORY_SCOPE_AGENT);
        unsigned a = __hip_atomic_fetch_add(&sync[0], 1u, __ATOMIC_ACQ_REL, __HIP_MEMORY_SCOPE_AGENT);
        if (a == (unsigned)(NWG - 1)) {
            __hip_atomic_store(&sync[0], 0u, __ATOMIC_RELAXED, __HIP_MEMORY_SCOPE_AGENT);
            __hip_atomic_fetch_add(&sync[1], 1u, __ATOMIC_RELEASE, __HIP_MEMORY_SCOPE_AGENT);
        } else {
            while (__hip_atomic_load(&sync[1], __ATOMIC_ACQUIRE, __HIP_MEMORY_SCOPE_AGENT) == g) {
                __builtin_amdgcn_s_sleep(2);
            }
        }
    }
    __syncthreads();
}

// ---------------- prep kernels ----------------
__global__ void init_sync_kernel(unsigned int* sync) {
    if (threadIdx.x < 2) sync[threadIdx.x] = 0u;
}

// X f32 -> bf16, streaming, 8 elements/thread
__global__ __launch_bounds__(256) void x_cvt_kernel(
    const float4* __restrict__ X4, uint4* __restrict__ Xb4) {
    size_t i = (size_t)blockIdx.x * 256 + threadIdx.x;
    float4 f0 = X4[2 * i];
    float4 f1 = X4[2 * i + 1];
    Pack8 p;
    p.us[0] = f2bf(f0.x); p.us[1] = f2bf(f0.y);
    p.us[2] = f2bf(f0.z); p.us[3] = f2bf(f0.w);
    p.us[4] = f2bf(f1.x); p.us[5] = f2bf(f1.y);
    p.us[6] = f2bf(f1.z); p.us[7] = f2bf(f1.w);
    Xb4[i] = p.q;
}

// W (K=512, N=2048) f32  ->  WT (N=2048, K=512) bf16
__global__ __launch_bounds__(256) void cvt_transpose_kernel(
    const float* __restrict__ W, unsigned short* __restrict__ WT) {
    int idx = blockIdx.x * 256 + threadIdx.x;   // over 2048*512
    int n = idx >> 9;
    int k = idx & 511;
    WT[idx] = f2bf(W[(size_t)k * G4 + n]);
}

// ---------------- phase 1: gx = X @ Wx + bx  (bf16 WMMA, f32 accum) ----------------
// Xbf: (S*B, 512) bf16; WxT: (2048, 512) bf16; gx: (S*B, 2048) f32
__global__ __launch_bounds__(256) void gx_gemm_kernel(
    const unsigned short* __restrict__ Xbf,
    const unsigned short* __restrict__ WxT,
    const float* __restrict__ bx,
    float* __restrict__ gx) {
    const int lane = threadIdx.x & 31;
    const int wave = threadIdx.x >> 5;
    const int gw   = blockIdx.x * 8 + wave;       // 0 .. 131071
    const int mtile = gw >> 5;                    // 0 .. 4095
    const int ng0   = (gw & 31) * 4;              // 4 N-tiles per wave

    const int half = lane >> 4;                   // 0/1
    const int mr   = lane & 15;
    const unsigned short* xrow = Xbf + (size_t)(mtile * 16 + mr) * INDIM;

    auto loadA = [&](int kt, BFrag& a) {
        const int kA = kt * 32 + half * 8;
        a.q[0] = *(const uint4*)(xrow + kA);
        a.q[1] = *(const uint4*)(xrow + kA + 16);
    };
    auto loadB = [&](int kt, int j, BFrag& b) {
        const int kB = kt * 32 + half * 8;
        const unsigned short* wrow =
            WxT + (size_t)((ng0 + j) * 16 + mr) * INDIM + kB;
        b.q[0] = *(const uint4*)(wrow);
        b.q[1] = *(const uint4*)(wrow + 16);
    };

    v8f acc[4];
#pragma unroll
    for (int j = 0; j < 4; ++j) acc[j] = zero8();

    // software-pipelined: fetch kt+1 while WMMAs of kt execute (fully unrolled)
    BFrag a, b[4];
    loadA(0, a);
#pragma unroll
    for (int j = 0; j < 4; ++j) loadB(0, j, b[j]);

#pragma unroll
    for (int kt = 0; kt < INDIM / 32; ++kt) {
        BFrag an, bn[4];
        if (kt + 1 < INDIM / 32) {
            loadA(kt + 1, an);
#pragma unroll
            for (int j = 0; j < 4; ++j) loadB(kt + 1, j, bn[j]);
        }
#pragma unroll
        for (int j = 0; j < 4; ++j)
            acc[j] = __builtin_amdgcn_wmma_f32_16x16x32_bf16(
                false, a.v, false, b[j].v, (short)0, acc[j], false, false);
        if (kt + 1 < INDIM / 32) {
            a = an;
#pragma unroll
            for (int j = 0; j < 4; ++j) b[j] = bn[j];
        }
    }

#pragma unroll
    for (int j = 0; j < 4; ++j) {
        const int n   = (ng0 + j) * 16 + mr;
        const float bxv = bx[n];
#pragma unroll
        for (int r = 0; r < 8; ++r) {
            const int m = mtile * 16 + half * 8 + r;
            gx[(size_t)m * G4 + n] = acc[j][r] + bxv;
        }
    }
}

// ---------------- phase 2: persistent recurrent kernel ----------------
// 16 WGs x 256 threads. WG wg owns hidden units [wg*32, wg*32+32) => 128 gate cols.
__global__ __launch_bounds__(256) void plstm_recurrent_kernel(
    const float* __restrict__ gx,            // (S*B, 2048), includes bx
    const unsigned short* __restrict__ WhT,  // (2048, 512) bf16
    const float* __restrict__ bh,            // (2048)
    const float* __restrict__ zc,            // (512)
    unsigned short* __restrict__ Hglob,      // ping-pong 2 x (32*512) bf16
    unsigned int* __restrict__ sync,         // [0]=counter [1]=generation
    float* __restrict__ out) {               // S*B*H  |  h_n B*H  |  c_n B*H
    __shared__ unsigned short Bs[128 * 520];   // Wh^T slice, 520-pitch (133 KB)
    __shared__ unsigned short Hs[32 * 520];    // full h_{t-1}, 520-pitch (33 KB)
    __shared__ float Gs[128 * 33];             // gate pre-activation exchange
    __shared__ float Cs[32 * 32];              // cell state (b, u)

    const int tid  = threadIdx.x;
    const int lane = tid & 31;
    const int wave = tid >> 5;
    const int wg   = blockIdx.x;
    const int hblk = wg * 32;

    const unsigned ldsBs = LDS_OFFSET(&Bs[0]);
    const unsigned ldsHs = LDS_OFFSET(&Hs[0]);

    // ---- TDM: pull Wh^T slice into Bs as a 3D tile ----
    // x: 512 elems/row, y: 32 rows (stride 512), z: 4 gate blocks (stride 512*512)
    if (tid == 0) {
        tdm_load_bf16(ldsBs, WhT + (size_t)hblk * INDIM,
                      /*tensor_d0=*/INDIM, /*tensor_d1=*/32, /*tensor_d2=*/4,
                      /*tile_d0=*/INDIM, /*tile_d1=*/32, /*tile_d2=*/4,
                      /*stride0=*/INDIM, /*stride1=*/(unsigned long long)HIDN * INDIM);
    }
    // zero cell state, zero h_{-1} slice in buffer 0
    for (int i = tid; i < 32 * 32; i += 256) Cs[i] = 0.f;
    for (int i = tid; i < 32 * 32; i += 256) {
        int b = i >> 5, u = i & 31;
        Hglob[b * HIDN + hblk + u] = 0;   // bf16 zero
    }

    // per-lane constants: each lane owns one gate column of its wave's N-tile
    const int jcol  = wave * 16 + (lane & 15);                    // 0..127
    const int Nglob = (jcol >> 5) * HIDN + hblk + (jcol & 31);    // column in 0..2047
    const float bhv = bh[Nglob];
    const int half  = lane >> 4;
    const int mr    = lane & 15;

    if (tid == 0) __builtin_amdgcn_s_wait_tensorcnt(0);
    __threadfence();
    gridBarrier(sync);

    for (int t = 0; t < S_LEN; ++t) {
        const unsigned short* Hsrc = Hglob + (size_t)(t & 1) * (BATCH * HIDN);
        unsigned short*       Hdst = Hglob + (size_t)((t + 1) & 1) * (BATCH * HIDN);

        // ---- TDM: pull full h_{t-1} (32x512 bf16) into Hs with 520 pitch ----
        if (tid == 0) {
            tdm_load_bf16(ldsHs, Hsrc,
                          /*tensor_d0=*/HIDN, /*tensor_d1=*/32, /*tensor_d2=*/0,
                          /*tile_d0=*/HIDN, /*tile_d1=*/32, /*tile_d2=*/0,
                          /*stride0=*/HIDN, /*stride1=*/0);
            __builtin_amdgcn_s_wait_tensorcnt(0);
        }
        __syncthreads();

        // prefetch this lane's gx column for the next timestep
        if (t + 1 < S_LEN)
            __builtin_prefetch(gx + ((size_t)(t + 1) * BATCH) * G4 + Nglob, 0, 0);

        // preload this step's gx values (overlaps with the WMMA loop below)
        const float* gxrow = gx + ((size_t)t * BATCH) * G4 + Nglob;
        float gxv0[8], gxv1[8];
#pragma unroll
        for (int r = 0; r < 8; ++r) {
            gxv0[r] = gxrow[(size_t)(half * 8 + r) * G4];
            gxv1[r] = gxrow[(size_t)(16 + half * 8 + r) * G4];
        }

        // ---- GEMM: h(32x512) @ WhT-slice -> this wave's 16 columns, M-tiles 0/1 ----
        v8f acc0 = zero8();
        v8f acc1 = zero8();
#pragma unroll 4
        for (int kt = 0; kt < INDIM / 32; ++kt) {
            const int kA = kt * 32 + half * 8;
            const int kB = kt * 32 + half * 8;
            BFrag a0, a1, b;
            a0.q[0] = *(const uint4*)(&Hs[mr * 520 + kA]);
            a0.q[1] = *(const uint4*)(&Hs[mr * 520 + kA + 16]);
            a1.q[0] = *(const uint4*)(&Hs[(16 + mr) * 520 + kA]);
            a1.q[1] = *(const uint4*)(&Hs[(16 + mr) * 520 + kA + 16]);
            b.q[0]  = *(const uint4*)(&Bs[jcol * 520 + kB]);
            b.q[1]  = *(const uint4*)(&Bs[jcol * 520 + kB + 16]);
            acc0 = __builtin_amdgcn_wmma_f32_16x16x32_bf16(
                false, a0.v, false, b.v, (short)0, acc0, false, false);
            acc1 = __builtin_amdgcn_wmma_f32_16x16x32_bf16(
                false, a1.v, false, b.v, (short)0, acc1, false, false);
        }

        // ---- add gx + bh, publish pre-activations to LDS ----
#pragma unroll
        for (int r = 0; r < 8; ++r) {
            int m0 = half * 8 + r;    // rows 0..15
            int m1 = 16 + m0;         // rows 16..31
            Gs[jcol * 33 + m0] = acc0[r] + bhv + gxv0[r];
            Gs[jcol * 33 + m1] = acc1[r] + bhv + gxv1[r];
        }
        __syncthreads();

        // ---- elementwise LSTM cell update for (b, u) pairs ----
        const bool lastStep = (t == S_LEN - 1);
        for (int p = tid; p < BATCH * 32; p += 256) {
            int b = p >> 5, u = p & 31;
            float iv = Gs[(0 * 32 + u) * 33 + b];
            float ov = Gs[(1 * 32 + u) * 33 + b];
            float zv = Gs[(2 * 32 + u) * 33 + b];
            float fv = Gs[(3 * 32 + u) * 33 + b];
            float i_ = sigf(iv);
            float o_ = sigf(ov);
            float f_ = sigf(fv);
            float cold = Cs[b * 32 + u];
            float z_ = tanhf(zv + zc[hblk + u] * cold);   // peephole into block input
            float c  = i_ * z_ + f_ * cold;
            float h  = o_ * tanhf(c);
            Cs[b * 32 + u] = c;
            out[((size_t)t * BATCH + b) * HIDN + hblk + u] = h;
            Hdst[b * HIDN + hblk + u] = f2bf(h);
            if (lastStep) {
                size_t base = (size_t)S_LEN * BATCH * HIDN;
                out[base + (size_t)b * HIDN + hblk + u] = h;                        // h_n
                out[base + (size_t)BATCH * HIDN + (size_t)b * HIDN + hblk + u] = c; // c_n
            }
        }
        __threadfence();
        gridBarrier(sync);
    }
}

// ---------------- launcher ----------------
extern "C" void kernel_launch(void* const* d_in, const int* in_sizes, int n_in,
                              void* d_out, int out_size, void* d_ws, size_t ws_size,
                              hipStream_t stream) {
    const float* X  = (const float*)d_in[0];   // (S, B, In)
    const float* Wx = (const float*)d_in[1];   // (In, 4H)
    const float* bx = (const float*)d_in[2];   // (4H)
    const float* Wh = (const float*)d_in[3];   // (H, 4H)
    const float* bh = (const float*)d_in[4];   // (4H)
    const float* zc = (const float*)d_in[5];   // (1, H)
    float* out = (float*)d_out;

    // workspace layout
    constexpr size_t GX_BYTES  = (size_t)S_LEN * BATCH * G4 * sizeof(float);      // 512 MiB
    constexpr size_t WT_BYTES  = (size_t)G4 * INDIM * sizeof(unsigned short);     // 2 MiB each
    constexpr size_t XB_BYTES  = (size_t)S_LEN * BATCH * INDIM * sizeof(unsigned short); // 64 MiB
    constexpr size_t HG_BYTES  = (size_t)2 * BATCH * HIDN * sizeof(unsigned short);

    char* ws = (char*)d_ws;
    float*          gxbuf = (float*)ws;
    unsigned short* WxT   = (unsigned short*)(ws + GX_BYTES);
    unsigned short* WhT   = (unsigned short*)(ws + GX_BYTES + WT_BYTES);
    unsigned short* Xbf   = (unsigned short*)(ws + GX_BYTES + 2 * WT_BYTES);
    unsigned short* Hglob = (unsigned short*)(ws + GX_BYTES + 2 * WT_BYTES + XB_BYTES);
    unsigned int*   sync  = (unsigned int*)(ws + GX_BYTES + 2 * WT_BYTES + XB_BYTES + HG_BYTES);

    init_sync_kernel<<<1, 64, 0, stream>>>(sync);
    x_cvt_kernel<<<(S_LEN * BATCH * INDIM) / (256 * 8), 256, 0, stream>>>(
        (const float4*)X, (uint4*)Xbf);
    cvt_transpose_kernel<<<(G4 * INDIM) / 256, 256, 0, stream>>>(Wx, WxT);
    cvt_transpose_kernel<<<(G4 * INDIM) / 256, 256, 0, stream>>>(Wh, WhT);
    // 4096 M-tiles x 128 N-tiles; each wave does 1 M-tile x 4 N-tiles; 8 waves/block
    gx_gemm_kernel<<<16384, 256, 0, stream>>>(Xbf, WxT, bx, gxbuf);
    plstm_recurrent_kernel<<<NWG, 256, 0, stream>>>(gxbuf, WhT, bh, zc, Hglob, sync, out);
}
